// k1_GNN_subconv_7842610283387
// MI455X (gfx1250) — compile-verified, matrix operand order checked by reading, not run on hardware
//
#include <hip/hip_runtime.h>
#include <hip/hip_bf16.h>
#include <cstdint>
#include <cstddef>

typedef __attribute__((ext_vector_type(16))) _Float16     v16h;
typedef __attribute__((ext_vector_type(8)))  float        v8f;
typedef __attribute__((ext_vector_type(8)))  unsigned int v8u;

#define K_EXT 160   // 128 hidden + 1 bias row (h=1) + 31 zero-pad rows -> 5 chunks of K=32

__device__ __forceinline__ float eluf(float x) {
    return x > 0.f ? x : (__expf(x) - 1.f);
}

// ---------------------------------------------------------------------------
// h_ext[e, 0:160] f16:  h[k<128] = relu(ea[e]@w1 + b1)[k];  h[128]=1;  rest 0
// ---------------------------------------------------------------------------
__global__ void k_edge_mlp(const float* __restrict__ ea,   // [E,5]
                           const float* __restrict__ w1,   // [5,128]
                           const float* __restrict__ b1,   // [128]
                           _Float16* __restrict__ h16,     // [E,160]
                           int E)
{
    int tid = blockIdx.x * blockDim.x + threadIdx.x;
    int e = tid / K_EXT, k = tid % K_EXT;
    if (e >= E) return;
    float v;
    if (k < 128) {
        v = b1[k];
        const float* ar = ea + (size_t)e * 5;
#pragma unroll
        for (int d = 0; d < 5; ++d) v += ar[d] * w1[d * 128 + k];
        v = fmaxf(v, 0.f);
    } else {
        v = (k == 128) ? 1.f : 0.f;
    }
    h16[(size_t)e * K_EXT + k] = (_Float16)v;
}

// ---------------------------------------------------------------------------
// Pack w2 [128, D] + b2 [D] (f32) into WMMA 16-bit B-matrix layout (f16).
// Layout per (column-tile nt, K-chunk kc): 32 lanes x 16 f16 contiguous.
// Lane L, element idx ee (v = ee/2, s = ee&1):
//   k = kc*32 + (v&4 ? 16 : 0) + (L/16)*8 + (v&3)*2 + s
//   n = nt*16 + (L&15)
// ---------------------------------------------------------------------------
__global__ void k_pack_w2(const float* __restrict__ w2,    // [128, D]
                          const float* __restrict__ b2,    // [D]
                          _Float16* __restrict__ w2p,      // [D/16 * 5 * 32 * 16]
                          int D)
{
    int tid = blockIdx.x * blockDim.x + threadIdx.x;
    if (tid >= K_EXT * D) return;
    int ee   = tid & 15;
    int lane = (tid >> 4) & 31;
    int kc   = (tid >> 9) % 5;
    int nt   = tid / 2560;
    int v = ee >> 1, s = ee & 1;
    int k = kc * 32 + ((v & 4) << 2) + ((lane >> 4) << 3) + (v & 3) * 2 + s;
    int n = nt * 16 + (lane & 15);
    float val = (k < 128) ? w2[(size_t)k * D + n] : (k == 128 ? b2[n] : 0.f);
    w2p[tid] = (_Float16)val;
}

// ---------------------------------------------------------------------------
// Fused: one wave owns 16 edges. For each (o-chunk, input feat i):
//   C = h_ext[16 edges] @ w2_ext[:, i*dout+oc*16 .. +16]   (5x WMMA, K=160)
//   macc[e, n] += x[src[e], i] * C[e, n]
// then scatter-add macc into agg[dst[e], :] with native f32 atomics.
// ---------------------------------------------------------------------------
__global__ void __launch_bounds__(32)
k_fused_msg(const _Float16* __restrict__ h16,  // [E,160]
            const _Float16* __restrict__ w2p,  // packed B
            const float*    __restrict__ xin,  // [N,din]
            const int*      __restrict__ ei,   // [2,E]
            float*          __restrict__ agg,  // [N,dout]
            int E, int din, int dout)
{
    __shared__ float xsh[16][65];   // padded rows
    __shared__ int   dsh[16];

    const int lane = threadIdx.x;
    const int e0   = blockIdx.x * 16;
    const int m    = lane & 15;
    const int half = lane >> 4;

    if (lane < 16) {   // gather src features + dst ids (reconverges before WMMA)
        int e = e0 + lane;
        int s = ei[e];
        dsh[lane] = ei[E + e];
        const float* xr = xin + (size_t)s * din;
        for (int i = 0; i < din; ++i) xsh[lane][i] = xr[i];
    }
    __syncthreads();

    // A tiles (h_ext) for our 16 edges, K = 5 chunks of 32, ISA 16-bit A layout
    v16h a[5];
    {
        const _Float16* hr = h16 + (size_t)(e0 + m) * K_EXT;
#pragma unroll
        for (int kc = 0; kc < 5; ++kc) {
            v8u au;
#pragma unroll
            for (int v = 0; v < 8; ++v) {
                int k = kc * 32 + ((v & 4) << 2) + (half << 3) + (v & 3) * 2;
                au[v] = *(const unsigned int*)(hr + k);  // (k, k+1) pair
            }
            a[kc] = __builtin_bit_cast(v16h, au);
        }
    }

    const int ntiles = dout >> 4;
    const int rbase  = half << 3;                 // C rows this lane holds
    for (int oc = 0; oc < ntiles; ++oc) {
        v8f macc = {0.f, 0.f, 0.f, 0.f, 0.f, 0.f, 0.f, 0.f};
        // packed tile nt = i*ntiles + oc; lane's data at ((nt*5+kc)*32+lane)*16
        const _Float16* bp = w2p + (size_t)oc * 2560 + (size_t)lane * 16;
        const size_t istep = (size_t)ntiles * 2560;
        for (int i = 0; i < din; ++i) {
            v8f c = {0.f, 0.f, 0.f, 0.f, 0.f, 0.f, 0.f, 0.f};
#pragma unroll
            for (int kc = 0; kc < 5; ++kc) {
                v16h b = *(const v16h*)(bp + kc * 512);
                c = __builtin_amdgcn_wmma_f32_16x16x32_f16(
                        false, a[kc], false, b, (short)0, c, false, false);
            }
#pragma unroll
            for (int r = 0; r < 8; ++r)
                macc[r] += xsh[rbase + r][i] * c[r];
            bp += istep;
        }
#pragma unroll
        for (int r = 0; r < 8; ++r) {
            int d = dsh[rbase + r];
            unsafeAtomicAdd(&agg[(size_t)d * dout + oc * 16 + m], macc[r]);
        }
    }
}

// ---------------------------------------------------------------------------
// out[n,o] = elu(agg[n,o] + x[n,:]@root[:,o] + bias[o])
// ---------------------------------------------------------------------------
__global__ void k_root_elu(const float* __restrict__ agg,
                           const float* __restrict__ xin,
                           const float* __restrict__ root,  // [din,dout]
                           const float* __restrict__ bias,  // [dout]
                           float* __restrict__ out,
                           int N, int din, int dout)
{
    int tid = blockIdx.x * blockDim.x + threadIdx.x;
    if (tid >= N * dout) return;
    int n = tid / dout, o = tid % dout;
    float v = agg[tid] + bias[o];
    const float* xr = xin + (size_t)n * din;
    for (int i = 0; i < din; ++i) v += xr[i] * root[i * dout + o];
    out[tid] = eluf(v);
}

// ---------------------------------------------------------------------------
// mean-pool (width fixed 64): scatter-add + counts, then divide
// ---------------------------------------------------------------------------
__global__ void k_pool_scatter(const float* __restrict__ xin, const int* __restrict__ seg,
                               float* __restrict__ pool, float* __restrict__ cnt, int N)
{
    int tid = blockIdx.x * blockDim.x + threadIdx.x;
    if (tid >= N * 64) return;
    int n = tid >> 6, o = tid & 63;
    int s = seg[n];
    unsafeAtomicAdd(&pool[(size_t)s * 64 + o], xin[tid]);
    if (o == 0) unsafeAtomicAdd(&cnt[s], 1.f);
}

__global__ void k_pool_div(float* __restrict__ pool, const float* __restrict__ cnt, int S)
{
    int tid = blockIdx.x * blockDim.x + threadIdx.x;
    if (tid >= S * 64) return;
    pool[tid] /= fmaxf(cnt[tid >> 6], 1.f);
}

// ---------------------------------------------------------------------------
// FC head: 64 graphs, 64->32->16->1 with ELU
// ---------------------------------------------------------------------------
__global__ void k_fc(const float* __restrict__ gx,
                     const float* __restrict__ w1, const float* __restrict__ b1,
                     const float* __restrict__ w2, const float* __restrict__ b2,
                     const float* __restrict__ w3, const float* __restrict__ b3,
                     float* __restrict__ out)
{
    int g = threadIdx.x;  // 64 threads
    float h0[64], h1[32], h2[16];
#pragma unroll
    for (int i = 0; i < 64; ++i) h0[i] = gx[g * 64 + i];
#pragma unroll
    for (int j = 0; j < 32; ++j) {
        float v = b1[j];
        for (int i = 0; i < 64; ++i) v += h0[i] * w1[i * 32 + j];
        h1[j] = eluf(v);
    }
#pragma unroll
    for (int j = 0; j < 16; ++j) {
        float v = b2[j];
        for (int i = 0; i < 32; ++i) v += h1[i] * w2[i * 16 + j];
        h2[j] = eluf(v);
    }
    float v = b3[0];
#pragma unroll
    for (int i = 0; i < 16; ++i) v += h2[i] * w3[i];
    out[g] = v;
}

// ---------------------------------------------------------------------------
// host driver
// ---------------------------------------------------------------------------
static void run_conv(hipStream_t stream, int E, int N, int din, int dout,
                     const float* xin, const int* ei, const float* ea,
                     const float* w1, const float* b1, const float* w2,
                     const float* b2, const float* root, const float* bias,
                     _Float16* h16, _Float16* w2p, float* agg, float* xout)
{
    int D = din * dout;
    k_edge_mlp<<<(E * K_EXT + 255) / 256, 256, 0, stream>>>(ea, w1, b1, h16, E);
    k_pack_w2 <<<(K_EXT * D + 255) / 256, 256, 0, stream>>>(w2, b2, w2p, D);
    hipMemsetAsync(agg, 0, (size_t)N * dout * sizeof(float), stream);
    k_fused_msg<<<E / 16, 32, 0, stream>>>(h16, w2p, xin, ei, agg, E, din, dout);
    k_root_elu <<<(N * dout + 255) / 256, 256, 0, stream>>>(agg, xin, root, bias, xout, N, din, dout);
}

extern "C" void kernel_launch(void* const* d_in, const int* in_sizes, int n_in,
                              void* d_out, int out_size, void* d_ws, size_t ws_size,
                              hipStream_t stream)
{
    const float* x    = (const float*)d_in[0];   // [20000,13]
    const int*   ei   = (const int*)  d_in[1];   // [2,40000]
    const float* ea   = (const float*)d_in[2];   // [40000,5]
    const int*   n2s  = (const int*)  d_in[3];   // [20000]
    const int*   oei  = (const int*)  d_in[4];   // [2,32000]
    const float* oea  = (const float*)d_in[5];   // [32000,5]
    const int*   s2g  = (const int*)  d_in[6];   // [2000]
    // sub_params base 7 (3 layers x 6), glob_params base 25 (2 x 6), fc base 37
    auto P = [&](int i) { return (const float*)d_in[i]; };

    // workspace carve-up (all 256B aligned)
    char* ws = (char*)d_ws;
    size_t off = 0;
    auto take = [&](size_t bytes) {
        void* p = ws + off;
        off = (off + bytes + 255) & ~(size_t)255;
        return p;
    };
    _Float16* h16  = (_Float16*)take((size_t)40000 * K_EXT * 2);
    _Float16* w2p  = (_Float16*)take((size_t)K_EXT * 4096 * 2);
    float* agg  = (float*)take((size_t)20000 * 64 * 4);
    float* bufA = (float*)take((size_t)20000 * 64 * 4);
    float* bufB = (float*)take((size_t)20000 * 64 * 4);
    float* subx = (float*)take((size_t)2000 * 64 * 4);
    float* cnt  = (float*)take((size_t)2048 * 4);
    float* gx   = (float*)take((size_t)64 * 64 * 4);
    float* cnt2 = (float*)take((size_t)64 * 4);

    const int E1 = 40000, N1 = 20000, E2 = 32000, N2 = 2000;

    // --- sub convs: 13->32, 32->64, 64->64 ---
    run_conv(stream, E1, N1, 13, 32, x,    ei, ea, P(7),  P(8),  P(9),  P(10), P(11), P(12), h16, w2p, agg, bufA);
    run_conv(stream, E1, N1, 32, 64, bufA, ei, ea, P(13), P(14), P(15), P(16), P(17), P(18), h16, w2p, agg, bufB);
    run_conv(stream, E1, N1, 64, 64, bufB, ei, ea, P(19), P(20), P(21), P(22), P(23), P(24), h16, w2p, agg, bufA);

    // --- pool nodes -> subgraphs ---
    hipMemsetAsync(subx, 0, (size_t)N2 * 64 * 4, stream);
    hipMemsetAsync(cnt,  0, (size_t)N2 * 4, stream);
    k_pool_scatter<<<(N1 * 64 + 255) / 256, 256, 0, stream>>>(bufA, n2s, subx, cnt, N1);
    k_pool_div    <<<(N2 * 64 + 255) / 256, 256, 0, stream>>>(subx, cnt, N2);

    // --- glob convs: 64->64, 64->64 ---
    run_conv(stream, E2, N2, 64, 64, subx, oei, oea, P(25), P(26), P(27), P(28), P(29), P(30), h16, w2p, agg, bufB);
    run_conv(stream, E2, N2, 64, 64, bufB, oei, oea, P(31), P(32), P(33), P(34), P(35), P(36), h16, w2p, agg, bufA);

    // --- pool subgraphs -> graphs ---
    hipMemsetAsync(gx,   0, (size_t)64 * 64 * 4, stream);
    hipMemsetAsync(cnt2, 0, (size_t)64 * 4, stream);
    k_pool_scatter<<<(N2 * 64 + 255) / 256, 256, 0, stream>>>(bufA, s2g, gx, cnt2, N2);
    k_pool_div    <<<(64 * 64 + 255) / 256, 256, 0, stream>>>(gx, cnt2, 64);

    // --- FC head ---
    k_fc<<<1, 64, 0, stream>>>(gx, P(37), P(38), P(39), P(40), P(41), P(42), (float*)d_out);
}
